// PrimitiveMixerBlock_9543417332027
// MI455X (gfx1250) — compile-verified, matrix operand order checked by reading, not run on hardware
//
#include <hip/hip_runtime.h>
#include <hip/hip_bf16.h>

// ---------------- types ----------------
typedef __attribute__((ext_vector_type(16))) __bf16 v16bf;
typedef __attribute__((ext_vector_type(8)))  float  v8f;
typedef __attribute__((ext_vector_type(4)))  unsigned int v4u;
typedef __attribute__((ext_vector_type(8)))  int    v8i;
typedef __attribute__((ext_vector_type(4)))  int    v4i;

union FragAB {
    v16bf v;
    uint4 q[2];
};

__device__ __forceinline__ unsigned short f2bf(float f) {
    unsigned u = __float_as_uint(f);
    unsigned r = u + 0x7FFFu + ((u >> 16) & 1u);   // round-to-nearest-even
    return (unsigned short)(r >> 16);
}

#if defined(__gfx1250__) && __has_builtin(__builtin_amdgcn_tensor_load_to_lds)
#define USE_TDM 1
#else
#define USE_TDM 0
#endif

// ---------------- LayerNorm -> bf16 h ----------------
__global__ __launch_bounds__(256)
void ln_kernel_9543(const float* __restrict__ x, const float* __restrict__ gamma,
                    const float* __restrict__ beta, unsigned short* __restrict__ h, int D) {
    const int row = blockIdx.x;
    const int t   = threadIdx.x;
    const float* xr = x + (size_t)row * D;

    float4 v = *(const float4*)&xr[t * 4];
    float s  = v.x + v.y + v.z + v.w;
    float s2 = v.x * v.x + v.y * v.y + v.z * v.z + v.w * v.w;

    __shared__ float rs_[8], rs2_[8];
    int lane = t & 31, wid = t >> 5;
    #pragma unroll
    for (int off = 16; off; off >>= 1) {
        s  += __shfl_down(s,  off, 32);
        s2 += __shfl_down(s2, off, 32);
    }
    if (lane == 0) { rs_[wid] = s; rs2_[wid] = s2; }
    __syncthreads();
    if (wid == 0) {
        float a = (lane < 8) ? rs_[lane]  : 0.f;
        float b = (lane < 8) ? rs2_[lane] : 0.f;
        #pragma unroll
        for (int off = 4; off; off >>= 1) {
            a += __shfl_down(a, off, 32);
            b += __shfl_down(b, off, 32);
        }
        if (lane == 0) { rs_[0] = a; rs2_[0] = b; }
    }
    __syncthreads();
    float mu  = rs_[0] / (float)D;
    float var = rs2_[0] / (float)D - mu * mu;
    float inv = rsqrtf(var + 1e-5f);

    float4 g  = *(const float4*)&gamma[t * 4];
    float4 be = *(const float4*)&beta[t * 4];
    ushort4 o;
    o.x = f2bf((v.x - mu) * inv * g.x + be.x);
    o.y = f2bf((v.y - mu) * inv * g.y + be.y);
    o.z = f2bf((v.z - mu) * inv * g.z + be.z);
    o.w = f2bf((v.w - mu) * inv * g.w + be.w);
    *(ushort4*)&h[(size_t)row * D + t * 4] = o;
}

// ---------------- f32 [K][N] -> bf16 transposed [N][K] ----------------
__global__ __launch_bounds__(256)
void cvtT_kernel_9543(const float* __restrict__ in, unsigned short* __restrict__ out,
                      int K, int N) {
    int i = blockIdx.x * 256 + threadIdx.x;     // i = k*N + n (coalesced read)
    if (i < K * N) {
        int k = i / N, n = i - k * N;
        out[(size_t)n * K + k] = f2bf(in[i]);
    }
}

// ---------------- TDM tile load: 128 rows x 32 bf16, row stride = K ----------------
#if USE_TDM
__device__ __forceinline__ void tdm_load_tile_9543(const unsigned short* gsrc,
                                                   unsigned lds_byte_off,
                                                   unsigned tensor_rows,
                                                   unsigned tensor_cols /* == stride */) {
    unsigned long long ga = (unsigned long long)(uintptr_t)gsrc;
    v4u g0;
    g0.x = 1u;                                             // count=1 (valid user D#)
    g0.y = lds_byte_off;                                   // lds_addr
    g0.z = (unsigned)(ga & 0xFFFFFFFFull);                 // global_addr[31:0]
    g0.w = (unsigned)((ga >> 32) & 0x01FFFFFFull)          // global_addr[56:32]
         | (2u << 30);                                     // type = 2 ("image")
    v8i g1;
    g1[0] = (int)(1u << 16);                               // data_size = 2 bytes
    g1[1] = (int)((tensor_cols & 0xFFFFu) << 16);          // tensor_dim0[15:0]
    g1[2] = (int)((tensor_cols >> 16) |
                  ((tensor_rows & 0xFFFFu) << 16));        // dim0[31:16] | dim1[15:0]
    g1[3] = (int)((tensor_rows >> 16) | (32u << 16));      // dim1[31:16] | tile_dim0=32
    g1[4] = 128;                                           // tile_dim1=128, tile_dim2=0
    g1[5] = (int)tensor_cols;                              // tensor_dim0_stride[31:0]
    g1[6] = 0;
    g1[7] = 0;
    v4i z4 = {0, 0, 0, 0};
    v8i z8 = {0, 0, 0, 0, 0, 0, 0, 0};
    __builtin_amdgcn_tensor_load_to_lds(g0, g1, z4, z4, z8, 0);
}
#endif

// ---------------- bf16 WMMA GEMM: C = A(MxK) * Bt(NxK)^T + bias (+x residual) ----------------
#define BM 128
#define BN 128
#define BK 32

template<bool RESID>
__global__ __launch_bounds__(256)
void gemm_bf16_9543(const unsigned short* __restrict__ A, const unsigned short* __restrict__ Bt,
                    const float* __restrict__ bias, const float* __restrict__ xres,
                    float* __restrict__ C, int M, int N, int K) {
    __shared__ alignas(16) unsigned short lds_a[2 * BM * BK];  // [buf][m][k]
    __shared__ alignas(16) unsigned short lds_b[2 * BN * BK];  // [buf][n][k]

    const int t    = threadIdx.x;
    const int bm   = blockIdx.y, bn = blockIdx.x;
    const int lane = t & 31, w = t >> 5;
    const int wm   = w >> 1, wn = w & 1;
    const int hlane = lane & 15, half = lane >> 4;
    const int nk   = K / BK;

    const unsigned short* Ab = A  + (size_t)(bm * BM) * K;
    const unsigned short* Bb = Bt + (size_t)(bn * BN) * K;

    v8f acc[2][4];
    #pragma unroll
    for (int i = 0; i < 2; ++i)
        #pragma unroll
        for (int j = 0; j < 4; ++j)
            acc[i][j] = (v8f){0.f, 0.f, 0.f, 0.f, 0.f, 0.f, 0.f, 0.f};

#if USE_TDM
    const unsigned ldsA0 = (unsigned)(uintptr_t)(void*)&lds_a[0];
    const unsigned ldsB0 = (unsigned)(uintptr_t)(void*)&lds_b[0];
    if (t < 32) {                       // wave 0 drives the TDM (EXEC ignored per-op)
        tdm_load_tile_9543(Ab + 0 * BK, ldsA0,               (unsigned)M, (unsigned)K);
        tdm_load_tile_9543(Bb + 0 * BK, ldsB0,               (unsigned)N, (unsigned)K);
        if (nk > 1) {
            tdm_load_tile_9543(Ab + 1 * BK, ldsA0 + BM * BK * 2, (unsigned)M, (unsigned)K);
            tdm_load_tile_9543(Bb + 1 * BK, ldsB0 + BN * BK * 2, (unsigned)N, (unsigned)K);
        }
    }
#else
    // synchronous fallback: stage buffer 0
    #pragma unroll
    for (int c = t; c < (BM * BK / 8); c += 256) {
        int row = c >> 2, kc = (c & 3) * 8;
        *(uint4*)&lds_a[row * BK + kc] = *(const uint4*)&Ab[(size_t)row * K + kc];
        *(uint4*)&lds_b[row * BK + kc] = *(const uint4*)&Bb[(size_t)row * K + kc];
    }
#endif

    for (int i = 0; i < nk; ++i) {
        const int buf = i & 1;
        const unsigned short* la = &lds_a[buf * BM * BK];
        const unsigned short* lb = &lds_b[buf * BN * BK];

#if USE_TDM
        if (t < 32) {
            if (i + 1 < nk) __builtin_amdgcn_s_wait_tensorcnt(2);  // pair i landed (in-order)
            else            __builtin_amdgcn_s_wait_tensorcnt(0);
        }
#endif
        __syncthreads();   // publish LDS tile i to all 8 waves

        FragAB fa[2], fb[4];
        #pragma unroll
        for (int ii = 0; ii < 2; ++ii) {
            int m = wm * 32 + ii * 16 + hlane;
            fa[ii].q[0] = *(const uint4*)&la[m * BK + half * 8];
            fa[ii].q[1] = *(const uint4*)&la[m * BK + 16 + half * 8];
        }
        #pragma unroll
        for (int jj = 0; jj < 4; ++jj) {
            int n = wn * 64 + jj * 16 + hlane;
            fb[jj].q[0] = *(const uint4*)&lb[n * BK + half * 8];
            fb[jj].q[1] = *(const uint4*)&lb[n * BK + 16 + half * 8];
        }
        #pragma unroll
        for (int ii = 0; ii < 2; ++ii)
            #pragma unroll
            for (int jj = 0; jj < 4; ++jj)
                acc[ii][jj] = __builtin_amdgcn_wmma_f32_16x16x32_bf16(
                    false, fa[ii].v, false, fb[jj].v, (short)0, acc[ii][jj], false, false);

        __syncthreads();   // all waves done reading buf before it is overwritten

#if USE_TDM
        if (t < 32 && (i + 2) < nk) {
            tdm_load_tile_9543(Ab + (size_t)(i + 2) * BK,
                               ldsA0 + buf * BM * BK * 2, (unsigned)M, (unsigned)K);
            tdm_load_tile_9543(Bb + (size_t)(i + 2) * BK,
                               ldsB0 + buf * BN * BK * 2, (unsigned)N, (unsigned)K);
        }
#else
        if (i + 1 < nk) {
            const int nb = (i + 1) & 1;
            #pragma unroll
            for (int c = t; c < (BM * BK / 8); c += 256) {
                int row = c >> 2, kc = (c & 3) * 8;
                *(uint4*)&lds_a[nb * BM * BK + row * BK + kc] =
                    *(const uint4*)&Ab[(size_t)row * K + (size_t)(i + 1) * BK + kc];
                *(uint4*)&lds_b[nb * BN * BK + row * BK + kc] =
                    *(const uint4*)&Bb[(size_t)row * K + (size_t)(i + 1) * BK + kc];
            }
        }
#endif
    }

    // epilogue: C/D layout -> row = half*8 + vgpr, col = lane&15
    #pragma unroll
    for (int i = 0; i < 2; ++i)
        #pragma unroll
        for (int j = 0; j < 4; ++j)
            #pragma unroll
            for (int vv = 0; vv < 8; ++vv) {
                int row = bm * BM + wm * 32 + i * 16 + half * 8 + vv;
                int col = bn * BN + wn * 64 + j * 16 + hlane;
                float r = acc[i][j][vv] + bias[col];
                if (RESID) r += xres[(size_t)row * N + col];
                C[(size_t)row * N + col] = r;
            }
}

// ---------------- sequential chaotic scan ----------------
__global__ __launch_bounds__(1024)
void scan_kernel_9543(const float* __restrict__ c, unsigned short* __restrict__ y,
                      int S, int D) {
    const int b = blockIdx.x;
    const int d = threadIdx.x;
    const int lane = d & 31, wid = d >> 5;
    __shared__ float zsh[1024];
    __shared__ float red[33];

    const float* cb = c + (size_t)b * S * D;
    unsigned short* yb = y + (size_t)b * S * D;

    float z = 0.f;
    for (int t = 0; t < S; ++t) {
        if (t + 1 < S) __builtin_prefetch(&cb[(size_t)(t + 1) * D + d], 0, 3);

        zsh[d] = z;
        __syncthreads();
        float re, im, zraw;
        if (d < 512) { re = zsh[d];       im = zsh[d + 512]; zraw = re * re - im * im; }
        else         { re = zsh[d - 512]; im = zsh[d];       zraw = 2.f * re * im; }
        zraw += cb[(size_t)t * D + d];

        float sq = zraw * zraw;
        #pragma unroll
        for (int off = 16; off; off >>= 1) sq += __shfl_down(sq, off, 32);
        if (lane == 0) red[wid] = sq;
        __syncthreads();
        if (wid == 0) {
            float v = red[lane];
            #pragma unroll
            for (int off = 16; off; off >>= 1) v += __shfl_down(v, off, 32);
            if (lane == 0) red[32] = v;
        }
        __syncthreads();
        float n = sqrtf(red[32] + 1e-12f);
        float s = 2.0f / fmaxf(n, 2.0f);
        z = zraw * s;
        yb[(size_t)t * D + d] = f2bf(z);
        __syncthreads();   // protect zsh/red before next iteration's writes
    }
}

// ---------------- launcher ----------------
extern "C" void kernel_launch(void* const* d_in, const int* in_sizes, int n_in,
                              void* d_out, int out_size, void* d_ws, size_t ws_size,
                              hipStream_t stream) {
    const float* x     = (const float*)d_in[0];
    const float* gamma = (const float*)d_in[1];
    const float* beta  = (const float*)d_in[2];
    const float* W_in  = (const float*)d_in[3];
    const float* b_in  = (const float*)d_in[4];
    const float* W_out = (const float*)d_in[5];
    const float* b_out = (const float*)d_in[6];

    const int Bb = 4, S = 4096, D = 1024;
    const int M = Bb * S;                         // 16384 rows

    char* ws = (char*)d_ws;
    unsigned short* h     = (unsigned short*)(ws);                    // 32 MB
    unsigned short* winT  = (unsigned short*)(ws + (size_t)33554432); //  2 MB  [N][K]
    unsigned short* woutT = (unsigned short*)(ws + (size_t)35651584); //  2 MB  [N][K]
    float*          cbuf  = (float*)        (ws + (size_t)37748736);  // 64 MB
    unsigned short* ybuf  = (unsigned short*)(ws + (size_t)104857600);// 32 MB

    // 1) LayerNorm -> bf16 h
    ln_kernel_9543<<<M, 256, 0, stream>>>(x, gamma, beta, h, D);

    // 2) weights -> bf16, transposed to [n][k] so GEMM B tiles are contiguous-K
    cvtT_kernel_9543<<<(D * D + 255) / 256, 256, 0, stream>>>(W_in,  winT,  D, D);
    cvtT_kernel_9543<<<(D * D + 255) / 256, 256, 0, stream>>>(W_out, woutT, D, D);

    // 3) c = h @ W_in + b_in   (bf16 WMMA, fp32 accum, TDM-staged tiles)
    dim3 grid(D / BN, M / BM);
    gemm_bf16_9543<false><<<grid, 256, 0, stream>>>(h, winT, b_in, nullptr, cbuf, M, D, D);

    // 4) sequential escape-clamped complex-square scan -> bf16 y
    scan_kernel_9543<<<Bb, 1024, 0, stream>>>(cbuf, ybuf, S, D);

    // 5) out = y @ W_out + b_out + x  (residual fused)
    gemm_bf16_9543<true><<<grid, 256, 0, stream>>>(ybuf, woutT, b_out, x, (float*)d_out, M, D, D);
}
